// LSTM_3590592660147
// MI455X (gfx1250) — matrix operand order repaired and checked by reading, not compiled, not run on hardware
//
#include <hip/hip_runtime.h>
#include <hip/hip_bf16.h>

typedef __attribute__((ext_vector_type(16))) __bf16 v16bf;
typedef __attribute__((ext_vector_type(8)))  __bf16 v8bf;
typedef __attribute__((ext_vector_type(8)))  float  v8f;
typedef __attribute__((ext_vector_type(4)))  float  v4f;

#define TT   512
#define BB   64
#define NINS 512
#define HIDS 512
#define KDIM 1024   // NIN + HID
#define NWG  32     // 512 hidden units / 16 units per WG
#define NTH  128    // 4 wave32 -> 4 M-tiles covering B=64

// Workspace layout:
//   [0..255]            : int barrier counter (rest padding)
//   [256 .. 256+128KB)  : bf16 h double-buffer, 2 x (64 x 512)
#define HBUF_OFF 256

__global__ void lstm_prep(unsigned int* ws) {
    const size_t n = (HBUF_OFF + 2u * BB * HIDS * sizeof(__bf16)) / 4;
    for (size_t i = blockIdx.x * blockDim.x + threadIdx.x; i < n;
         i += (size_t)gridDim.x * blockDim.x)
        ws[i] = 0u;
}

__global__ void __launch_bounds__(NTH, 1) lstm_persistent(
    const float* __restrict__ x,  const float* __restrict__ mask,
    const float* __restrict__ Wf, const float* __restrict__ bfv,
    const float* __restrict__ Wi, const float* __restrict__ biv,
    const float* __restrict__ Wc, const float* __restrict__ bcv,
    const float* __restrict__ Wo, const float* __restrict__ bov,
    float* __restrict__ out, int* __restrict__ bar, __bf16* __restrict__ hbuf)
{
    // Dynamic LDS: B-matrix fragments, [kb=32][gate=4][lane=32][16 bf16] = 128 KB
    extern __shared__ __bf16 sB[];

    const int tid   = threadIdx.x;
    const int wg    = blockIdx.x;      // 0..31 -> hidden-unit tile
    const int wave  = tid >> 5;        // 0..3  -> batch-row tile
    const int lane  = tid & 31;
    const int lhalf = lane >> 4;       // 0/1
    const int l16   = lane & 15;
    const int ubase = wg * 16;
    const int rbase = wave * 16;

    // gate order matches reference z layout: f, i, o, c~
    const float* Wsrc[4] = { Wf, Wi, Wo, Wc };
    const float* bsrc[4] = { bfv, biv, bov, bcv };

    // ---- stage weight B-fragments into LDS (f32 -> bf16), once ----
    // B layout (32x16 bf16): lane l -> N = l%16, K = kb*32 + (l>=16?8:0) + {0..7, 16..23}
    for (int e = tid; e < 32 * 4 * 32; e += NTH) {
        const int kb = e >> 7;
        const int g  = (e >> 5) & 3;
        const int ln = e & 31;
        const int n  = ubase + (ln & 15);
        const int ks = kb * 32 + ((ln >> 4) ? 8 : 0);
        const float* src = Wsrc[g] + (size_t)n * KDIM + ks;
        __bf16* dst = sB + (size_t)e * 16;
        #pragma unroll
        for (int j = 0; j < 16; ++j) dst[j] = (__bf16)src[j + (j >= 8 ? 8 : 0)];
    }
    __syncthreads();

    float biasv[4];
    #pragma unroll
    for (int g = 0; g < 4; ++g) biasv[g] = bsrc[g][ubase + l16];

    // per-wave persistent state tiles in WMMA C/D layout:
    // VGPR r, lane: row R = rbase + r + lhalf*8, unit U = ubase + l16
    v8f cstate = {0.f,0.f,0.f,0.f,0.f,0.f,0.f,0.f};
    v8f hprev  = {0.f,0.f,0.f,0.f,0.f,0.f,0.f,0.f};

    const size_t TBH = (size_t)TT * BB * HIDS;

    for (int t = 0; t < TT; ++t) {
        const __bf16* hrd = hbuf + (size_t)(t & 1) * (BB * HIDS);
        __bf16*       hwr = hbuf + (size_t)((t + 1) & 1) * (BB * HIDS);

        v8f acc[4];
        #pragma unroll
        for (int g = 0; g < 4; ++g) acc[g] = (v8f){0.f,0.f,0.f,0.f,0.f,0.f,0.f,0.f};

        // ---- K = 0..511 : recurrent part, A from bf16 h buffer ----
        // A layout (16x32 bf16): lane -> M = l16, K = kb*32 + lhalf*8 + {0..7, 16..23}
        const __bf16* aRowH = hrd + (size_t)(rbase + l16) * HIDS + lhalf * 8;
        for (int kb = 0; kb < 16; ++kb) {
            v8bf lo = *(const v8bf*)(aRowH + kb * 32);
            v8bf hi = *(const v8bf*)(aRowH + kb * 32 + 16);
            v16bf a = __builtin_shufflevector(lo, hi,
                      0,1,2,3,4,5,6,7, 8,9,10,11,12,13,14,15);
            #pragma unroll
            for (int g = 0; g < 4; ++g) {
                const v16bf b = *(const v16bf*)(sB + ((size_t)((kb * 4 + g) * 32 + lane)) * 16);
                acc[g] = __builtin_amdgcn_wmma_f32_16x16x32_bf16(
                             false, a, false, b, (short)0, acc[g], false, false);
            }
        }

        // ---- K = 512..1023 : input part, A from f32 x, converted in flight ----
        const float* xRow = x + ((size_t)t * BB + rbase + l16) * NINS + lhalf * 8;
        for (int kb = 0; kb < 16; ++kb) {
            const float* p = xRow + kb * 32;
            v4f p0 = *(const v4f*)(p);
            v4f p1 = *(const v4f*)(p + 4);
            v4f p2 = *(const v4f*)(p + 16);
            v4f p3 = *(const v4f*)(p + 20);
            v16bf a;
            #pragma unroll
            for (int j = 0; j < 4; ++j) {
                a[j]      = (__bf16)p0[j];
                a[4 + j]  = (__bf16)p1[j];
                a[8 + j]  = (__bf16)p2[j];
                a[12 + j] = (__bf16)p3[j];
            }
            #pragma unroll
            for (int g = 0; g < 4; ++g) {
                const v16bf b = *(const v16bf*)(sB + ((size_t)(((kb + 16) * 4 + g) * 32 + lane)) * 16);
                acc[g] = __builtin_amdgcn_wmma_f32_16x16x32_bf16(
                             false, a, false, b, (short)0, acc[g], false, false);
            }
        }

        // ---- gates + state update, all in C/D register layout ----
        const float* mp = mask + (size_t)t * BB + rbase + lhalf * 8;
        v4f m0 = *(const v4f*)(mp);
        v4f m1 = *(const v4f*)(mp + 4);
        const size_t obase = (size_t)t * BB * HIDS;
        const int U = ubase + l16;

        #pragma unroll
        for (int r = 0; r < 8; ++r) {
            const float zf = acc[0][r] + biasv[0];
            const float zi = acc[1][r] + biasv[1];
            const float zo = acc[2][r] + biasv[2];
            const float zc = acc[3][r] + biasv[3];
            const float fg = 1.f / (1.f + __expf(-zf));
            const float ig = 1.f / (1.f + __expf(-zi));
            const float og = 1.f / (1.f + __expf(-zo));
            const float ct = tanhf(zc);
            const float m  = (r < 4) ? m0[r & 3] : m1[r & 3];
            float cn = fg * cstate[r] + ig * ct;
            cn = m * cn + (1.f - m) * cstate[r];
            float hn = og * tanhf(cn);
            hn = m * hn + (1.f - m) * hprev[r];
            cstate[r] = cn;
            hprev[r]  = hn;
            const int R = rbase + r + lhalf * 8;
            out[obase + (size_t)R * HIDS + U]       = hn;   // h_hist
            out[TBH + obase + (size_t)R * HIDS + U] = ig;   // i_hist
            hwr[(size_t)R * HIDS + U] = (__bf16)hn;         // next-step A input
        }

        // ---- device-wide phase barrier (monotonic counter, reset by prep) ----
        __threadfence();
        __syncthreads();
        if (tid == 0) {
            atomicAdd(bar, 1);
            const int target = NWG * (t + 1);
            while (__hip_atomic_load(bar, __ATOMIC_RELAXED, __HIP_MEMORY_SCOPE_AGENT) < target)
                __builtin_amdgcn_s_sleep(2);
        }
        __syncthreads();
        __threadfence();
    }
}

extern "C" void kernel_launch(void* const* d_in, const int* in_sizes, int n_in,
                              void* d_out, int out_size, void* d_ws, size_t ws_size,
                              hipStream_t stream) {
    const float* x    = (const float*)d_in[0];
    const float* mask = (const float*)d_in[1];
    const float* Wf   = (const float*)d_in[2];
    const float* bf_  = (const float*)d_in[3];
    const float* Wi   = (const float*)d_in[4];
    const float* bi_  = (const float*)d_in[5];
    const float* Wc   = (const float*)d_in[6];
    const float* bc_  = (const float*)d_in[7];
    const float* Wo   = (const float*)d_in[8];
    const float* bo_  = (const float*)d_in[9];
    float* out = (float*)d_out;

    int*    bar  = (int*)d_ws;
    __bf16* hbuf = (__bf16*)((char*)d_ws + HBUF_OFF);

    lstm_prep<<<64, 256, 0, stream>>>((unsigned int*)d_ws);

    const size_t ldsBytes = 32u * 4u * 32u * 16u * sizeof(__bf16); // 128 KB B-fragments
    lstm_persistent<<<NWG, NTH, ldsBytes, stream>>>(
        x, mask, Wf, bf_, Wi, bi_, Wc, bc_, Wo, bo_, out, bar, hbuf);
}
